// MultiHeadSelfAttention_88759794139597
// MI455X (gfx1250) — matrix operand order repaired and checked by reading, not compiled, not run on hardware
//
#include <hip/hip_runtime.h>
#include <hip/hip_bf16.h>

// ---------------------------------------------------------------------------
// MHSA + T5 relative bias for gfx1250 (MI455X). All GEMMs on
// v_wmma_f32_16x16x32_bf16 (wave32). bf16 operands staged via
// GLOBAL_LOAD_ASYNC_TO_LDS (ASYNCcnt) where no dtype conversion is needed.
// Shapes: B=2,S=2048,D=1024,H=16,hd=64.
// ---------------------------------------------------------------------------

typedef __bf16 bf16_t;
typedef __attribute__((ext_vector_type(16))) __bf16 v16bf;
typedef __attribute__((ext_vector_type(8)))  __bf16 v8bf;
typedef __attribute__((ext_vector_type(8)))  float  v8f;

union Frag { v16bf v; v8bf h[2]; };

__device__ __forceinline__ v8f wmma_bf16(const Frag& a, const Frag& b, v8f c) {
  return __builtin_amdgcn_wmma_f32_16x16x32_bf16(
      /*neg_a=*/false, a.v, /*neg_b=*/false, b.v,
      /*c_mod=*/(short)0, c, /*reuse_a=*/false, /*reuse_b=*/false);
}

// A fragment (16x32, MxK): lane half h picks 16 contiguous K {h*16..h*16+15}
__device__ __forceinline__ Frag load_a_frag(const bf16_t* rowptr, int h) {
  Frag f;
  f.h[0] = *(const v8bf*)(rowptr + h * 16);
  f.h[1] = *(const v8bf*)(rowptr + h * 16 + 8);
  return f;
}
// B fragment (32x16, KxN): lane holds col n = lane&15; 16 contiguous K at
// k0 + (lane>>4)*16. colptr -> element (n, k0 + h*16).
__device__ __forceinline__ Frag load_b_frag(const bf16_t* colptr) {
  Frag f;
  f.h[0] = *(const v8bf*)(colptr);
  f.h[1] = *(const v8bf*)(colptr + 8);
  return f;
}

// Async bulk copy: 16 bytes/lane, global -> LDS, tracked by ASYNCcnt.
__device__ __forceinline__ void async_b128(void* lds_ptr, const void* gptr) {
  unsigned lds_off = (unsigned)(size_t)lds_ptr;   // low 32 bits = LDS offset
  asm volatile("global_load_async_to_lds_b128 %0, %1, off"
               :: "v"(lds_off), "v"(gptr) : "memory");
}
__device__ __forceinline__ void async_wait0() {
  asm volatile("s_wait_asynccnt 0" ::: "memory");
}

#define S_LEN   2048
#define D_MODEL 1024
#define NHEAD   16
#define HDIM    64
#define MROWS   4096          // B*S
#define PCOLS   272           // 257 rel slots padded to 17 wmma tiles
#define ZEROS8  {0.f,0.f,0.f,0.f,0.f,0.f,0.f,0.f}

// ---------------------------------------------------------------------------
// rel_emb f32 [257,64] -> bf16 [272,64] zero-padded
// ---------------------------------------------------------------------------
__global__ void k_cvt_rel(const float* __restrict__ rel, bf16_t* __restrict__ relp) {
  int i = blockIdx.x * 256 + threadIdx.x;
  if (i < PCOLS * HDIM) {
    int r = i >> 6;
    relp[i] = (r < 257) ? (bf16_t)rel[i] : (bf16_t)0.0f;
  }
}

// ---------------------------------------------------------------------------
// Generic C = A[M,K] @ B[N,K]^T  (both f32 in global, bf16-staged in LDS).
// Block tile 128x64, 8 waves as 4x2 grid of 32x32 wave tiles.
// out_mode 0: f32 row-major + bias
// out_mode 1: bf16 [b,h,s,hd]      (Q,K heads layout)
// out_mode 2: bf16 [b,h,hd,s]      (V transposed for contiguous-K B frags)
// ---------------------------------------------------------------------------
__global__ __launch_bounds__(256)
void k_gemm_nt(const float* __restrict__ A, const float* __restrict__ Bw,
               int M, int N, int K,
               float* __restrict__ Cf, const float* __restrict__ bias,
               bf16_t* __restrict__ Cb, int out_mode) {
  __shared__ __align__(16) bf16_t asub[128 * 32];
  __shared__ __align__(16) bf16_t bsub[64 * 32];
  const int tid  = threadIdx.x;
  const int wave = tid >> 5, lane = tid & 31;
  const int h = lane >> 4, lm = lane & 15;
  const int wr = wave & 3, wc = wave >> 2;
  const int row0 = blockIdx.x * 128;
  const int col0 = blockIdx.y * 64;

  v8f acc[2][2] = {{ZEROS8, ZEROS8}, {ZEROS8, ZEROS8}};

  for (int k0 = 0; k0 < K; k0 += 32) {
    if (k0 + 32 < K)   // pull next A chunk toward L2/L0 (global_prefetch_b8)
      __builtin_prefetch(&A[(size_t)(row0 + (tid >> 1)) * K + k0 + 32], 0, 1);
    for (int i = tid; i < 128 * 32; i += 256) {
      int r = i >> 5, c = i & 31;
      asub[i] = (bf16_t)A[(size_t)(row0 + r) * K + k0 + c];
    }
    for (int i = tid; i < 64 * 32; i += 256) {
      int r = i >> 5, c = i & 31;
      bsub[i] = (bf16_t)Bw[(size_t)(col0 + r) * K + k0 + c];
    }
    __syncthreads();
    #pragma unroll
    for (int i = 0; i < 2; ++i) {
      Frag a = load_a_frag(&asub[(wr * 32 + i * 16 + lm) * 32], h);
      #pragma unroll
      for (int j = 0; j < 2; ++j) {
        Frag b = load_b_frag(&bsub[(wc * 32 + j * 16 + lm) * 32 + h * 16]);
        acc[i][j] = wmma_bf16(a, b, acc[i][j]);
      }
    }
    __syncthreads();
  }

  #pragma unroll
  for (int i = 0; i < 2; ++i)
    #pragma unroll
    for (int j = 0; j < 2; ++j)
      #pragma unroll
      for (int e = 0; e < 8; ++e) {
        int m = row0 + wr * 32 + i * 16 + e + 8 * h;   // D layout: M=e(+8 hi half)
        int n = col0 + wc * 32 + j * 16 + lm;          // N = lane&15
        float v = acc[i][j][e];
        if (out_mode == 0) {
          if (bias) v += bias[n];
          Cf[(size_t)m * N + n] = v;
        } else {
          int b = m >> 11, s = m & (S_LEN - 1);
          int hh = n >> 6, d = n & (HDIM - 1);
          size_t off = (out_mode == 1)
              ? ((((size_t)b * NHEAD + hh) * S_LEN + s) * HDIM + d)
              : ((((size_t)b * NHEAD + hh) * HDIM + d) * S_LEN + s);
          Cb[off] = (bf16_t)v;
        }
      }
}

// ---------------------------------------------------------------------------
// P[m, r] = Q[m, :64] . relp[r, :64]   (m = bh*2048+s, r in [0,272))
// Fragments straight from global (K=64, rel stays L0-resident).
// ---------------------------------------------------------------------------
__global__ __launch_bounds__(256)
void k_pgemm(const bf16_t* __restrict__ Q, const bf16_t* __restrict__ relp,
             float* __restrict__ P) {
  const int tid = threadIdx.x, wave = tid >> 5, lane = tid & 31;
  const int h = lane >> 4, lm = lane & 15;
  const int row0 = blockIdx.x * 128 + wave * 16;

  Frag a[2];
  #pragma unroll
  for (int kc = 0; kc < 2; ++kc)
    a[kc] = load_a_frag(Q + (size_t)(row0 + lm) * HDIM + kc * 32, h);

  for (int ct = 0; ct < PCOLS / 16; ++ct) {
    v8f acc = ZEROS8;
    #pragma unroll
    for (int kc = 0; kc < 2; ++kc) {
      Frag b = load_b_frag(relp + (size_t)(ct * 16 + lm) * HDIM + kc * 32 + h * 16);
      acc = wmma_bf16(a[kc], b, acc);
    }
    #pragma unroll
    for (int e = 0; e < 8; ++e)
      P[(size_t)(row0 + e + 8 * h) * PCOLS + ct * 16 + lm] = acc[e];
  }
}

// ---------------------------------------------------------------------------
// Fused scores + bias-gather + mask + softmax for one (b,h, 16-row q block).
// K tiles arrive via async global->LDS copies (ASYNCcnt); 128KB LDS score
// strip is legal only on CDNA5 (320KB/WGP).
// ---------------------------------------------------------------------------
__global__ __launch_bounds__(256)
void k_scores(const bf16_t* __restrict__ Qb, const bf16_t* __restrict__ Kb,
              const float* __restrict__ P, const int* __restrict__ mask,
              float* __restrict__ attn) {
  __shared__ __align__(16) bf16_t qtile[16 * HDIM];     // 2 KB
  __shared__ __align__(16) bf16_t ktile[128 * HDIM];    // 16 KB (async dest)
  __shared__ float sc[16 * S_LEN];                      // 128 KB
  __shared__ float red[16 * 16];
  __shared__ float rowinv[16];

  const int tid = threadIdx.x, wave = tid >> 5, lane = tid & 31;
  const int h = lane >> 4, lm = lane & 15;
  const int bh = blockIdx.x >> 7;         // 0..31
  const int q0 = (blockIdx.x & 127) * 16;
  const int b  = bh >> 4;

  const size_t qbase = ((size_t)bh * S_LEN + q0) * HDIM;
  for (int i = tid; i < 16 * HDIM; i += 256) qtile[i] = Qb[qbase + i];

  // --- scores = Q K^T: 16 blocks of 128 K-rows; one 16-col tile per wave ---
  for (int cb = 0; cb < S_LEN / 128; ++cb) {
    const bf16_t* ksrc = Kb + ((size_t)bh * S_LEN + cb * 128) * HDIM;
    #pragma unroll
    for (int t = 0; t < 4; ++t) {        // 256 thr * 4 * 16B = 16 KB tile
      int e8 = (tid + t * 256) * 8;
      async_b128(&ktile[e8], ksrc + e8);
    }
    async_wait0();
    __syncthreads();                      // qtile (cb==0) + all waves' ktile

    const int kcol0 = cb * 128 + wave * 16;
    v8f acc = ZEROS8;
    #pragma unroll
    for (int kc = 0; kc < 2; ++kc) {
      Frag a = load_a_frag(qtile + lm * HDIM + kc * 32, h);
      Frag bfrag = load_b_frag(&ktile[(wave * 16 + lm) * HDIM + kc * 32 + h * 16]);
      acc = wmma_bf16(a, bfrag, acc);
    }
    #pragma unroll
    for (int e = 0; e < 8; ++e)
      sc[(e + 8 * h) * S_LEN + kcol0 + lm] = acc[e];
    __syncthreads();                      // all reads done before tile reuse
  }

  // --- scale + rel-bias gather + mask + softmax (16 threads per row) ---
  const int r = tid >> 4, sub = tid & 15;
  const int q = q0 + r;
  const float* Prow = P + ((size_t)bh * S_LEN + q) * PCOLS;
  const int*   mrow = mask + ((size_t)b * S_LEN + q) * S_LEN;
  const float inv_scale = 0.125f;         // 1/sqrt(64)

  float mx = -3.0e38f;
  for (int k = sub; k < S_LEN; k += 16) {
    int dlt = k - q;
    dlt = dlt < -128 ? -128 : (dlt > 128 ? 128 : dlt);
    float v = sc[r * S_LEN + k] * inv_scale + Prow[dlt + 128];
    if (mrow[k] == 0) v = -1.0e9f;
    sc[r * S_LEN + k] = v;
    mx = fmaxf(mx, v);
  }
  red[r * 16 + sub] = mx;
  __syncthreads();
  float rm = red[r * 16];
  #pragma unroll
  for (int j = 1; j < 16; ++j) rm = fmaxf(rm, red[r * 16 + j]);
  __syncthreads();

  float s = 0.f;
  for (int k = sub; k < S_LEN; k += 16) {
    float e = __expf(sc[r * S_LEN + k] - rm);
    sc[r * S_LEN + k] = e;
    s += e;
  }
  red[r * 16 + sub] = s;
  __syncthreads();
  if (sub == 0) {
    float rs = 0.f;
    #pragma unroll
    for (int j = 0; j < 16; ++j) rs += red[r * 16 + j];
    rowinv[r] = 1.0f / rs;
  }
  __syncthreads();

  // --- coalesced attn write: sc strip is exactly attn rows q0..q0+15 ---
  const size_t abase = ((size_t)bh * S_LEN + q0) * S_LEN;
  for (int i = tid; i < 16 * S_LEN; i += 256)
    attn[abase + i] = sc[i] * rowinv[i >> 11];
}

// ---------------------------------------------------------------------------
// ctx = attn @ V per (b,h): M=2048 (q), N=64 (d), K=2048 (k). attn f32 -> bf16
// via VGPR convert; V (already bf16, pre-transposed [b,h,d,s]) arrives via
// async global->LDS copies. Output ctx f32 [b, s, h*64+d].
// ---------------------------------------------------------------------------
__global__ __launch_bounds__(256)
void k_ctx(const float* __restrict__ attn, const bf16_t* __restrict__ Vt,
           float* __restrict__ ctx) {
  __shared__ __align__(16) bf16_t asub[128 * 32];   // 8 KB (converted attn)
  __shared__ __align__(16) bf16_t bsub[64 * 32];    // 4 KB (async dest)
  const int tid = threadIdx.x, wave = tid >> 5, lane = tid & 31;
  const int h = lane >> 4, lm = lane & 15;
  const int wr = wave & 3, wc = wave >> 2;
  const int bh = blockIdx.x >> 4;
  const int row0 = (blockIdx.x & 15) * 128;
  const int b = bh >> 4, hh = bh & 15;

  const float*  Arow = attn + ((size_t)bh * S_LEN + row0) * S_LEN;
  const bf16_t* Vb   = Vt + (size_t)bh * HDIM * S_LEN;

  v8f acc[2][2] = {{ZEROS8, ZEROS8}, {ZEROS8, ZEROS8}};

  for (int k0 = 0; k0 < S_LEN; k0 += 32) {
    {   // async stage bsub[d][c] = Vt[d][k0+c], 64x32 bf16 = 4 KB
      int d = tid >> 2, c8 = (tid & 3) * 8;
      async_b128(&bsub[d * 32 + c8], Vb + (size_t)d * S_LEN + k0 + c8);
    }
    for (int i = tid; i < 128 * 32; i += 256) {
      int r = i >> 5, c = i & 31;
      asub[i] = (bf16_t)Arow[(size_t)r * S_LEN + k0 + c];
    }
    async_wait0();
    __syncthreads();
    #pragma unroll
    for (int i = 0; i < 2; ++i) {
      Frag a = load_a_frag(&asub[(wr * 32 + i * 16 + lm) * 32], h);
      #pragma unroll
      for (int j = 0; j < 2; ++j) {
        Frag bfrag = load_b_frag(&bsub[(wc * 32 + j * 16 + lm) * 32 + h * 16]);
        acc[i][j] = wmma_bf16(a, bfrag, acc[i][j]);
      }
    }
    __syncthreads();
  }

  #pragma unroll
  for (int i = 0; i < 2; ++i)
    #pragma unroll
    for (int j = 0; j < 2; ++j)
      #pragma unroll
      for (int e = 0; e < 8; ++e) {
        int m = row0 + wr * 32 + i * 16 + e + 8 * h;
        int d = wc * 32 + j * 16 + lm;
        ctx[(((size_t)b * S_LEN + m) * NHEAD + hh) * HDIM + d] = acc[i][j][e];
      }
}

// ---------------------------------------------------------------------------
extern "C" void kernel_launch(void* const* d_in, const int* in_sizes, int n_in,
                              void* d_out, int out_size, void* d_ws, size_t ws_size,
                              hipStream_t stream) {
  const float* x     = (const float*)d_in[0];
  const int*   mask  = (const int*)  d_in[1];
  const float* w_q   = (const float*)d_in[2];
  const float* w_k   = (const float*)d_in[3];
  const float* w_v   = (const float*)d_in[4];
  const float* w_o_w = (const float*)d_in[5];
  const float* w_o_b = (const float*)d_in[6];
  const float* rel   = (const float*)d_in[7];

  float* out  = (float*)d_out;                         // [B,S,D]
  float* attn = out + (size_t)2 * S_LEN * D_MODEL;     // [B,H,S,S]

  // workspace carve (~108 MB)
  char* p = (char*)d_ws;
  bf16_t* Qb   = (bf16_t*)p;  p += (size_t)MROWS * D_MODEL * 2;   // 8 MB
  bf16_t* Kb   = (bf16_t*)p;  p += (size_t)MROWS * D_MODEL * 2;   // 8 MB
  bf16_t* Vt   = (bf16_t*)p;  p += (size_t)MROWS * D_MODEL * 2;   // 8 MB
  bf16_t* relp = (bf16_t*)p;  p += 65536;                          // 64 KB
  float*  P    = (float*)p;   p += (size_t)(MROWS * NHEAD) * PCOLS * 4; // 68 MB
  float*  ctx  = (float*)p;                                        // 16 MB

  dim3 gProj(MROWS / 128, D_MODEL / 64);   // 32 x 16

  k_cvt_rel<<<(PCOLS * HDIM + 255) / 256, 256, 0, stream>>>(rel, relp);

  k_gemm_nt<<<gProj, 256, 0, stream>>>(x, w_q, MROWS, D_MODEL, D_MODEL,
                                       nullptr, nullptr, Qb, 1);
  k_gemm_nt<<<gProj, 256, 0, stream>>>(x, w_k, MROWS, D_MODEL, D_MODEL,
                                       nullptr, nullptr, Kb, 1);
  k_gemm_nt<<<gProj, 256, 0, stream>>>(x, w_v, MROWS, D_MODEL, D_MODEL,
                                       nullptr, nullptr, Vt, 2);

  k_pgemm<<<(MROWS * NHEAD) / 128, 256, 0, stream>>>(Qb, relp, P);

  k_scores<<<2 * NHEAD * (S_LEN / 16), 256, 0, stream>>>(Qb, Kb, P, mask, attn);

  k_ctx<<<2 * NHEAD * (S_LEN / 128), 256, 0, stream>>>(attn, Vt, ctx);

  k_gemm_nt<<<gProj, 256, 0, stream>>>(ctx, w_o_w, MROWS, D_MODEL, D_MODEL,
                                       out, w_o_b, nullptr, 0);
}
